// Linear_TT_13322988552198
// MI455X (gfx1250) — 8-bit matrix operand order repaired and checked by reading, compiled, not run on hardware
//
#include <hip/hip_runtime.h>

typedef __attribute__((ext_vector_type(8))) int   v8i;
typedef __attribute__((ext_vector_type(4))) float v4f;

#define MAXQ 127.0f
#define MINQ -128.0f

__device__ __forceinline__ float fq(float x, float s) {
    float r = rintf(x / s);
    r = fminf(fmaxf(r, MINQ), MAXQ);
    return s * r;
}
__device__ __forceinline__ int q8(float x, float inv_s) {
    float r = rintf(x * inv_s);
    r = fminf(fmaxf(r, MINQ), MAXQ);
    return (int)r;
}

__global__ __launch_bounds__(256)
void prep_kernel(const float* __restrict__ core0, const float* __restrict__ core1,
                 const float* __restrict__ core2, const float* __restrict__ core3,
                 const float* sc0, const float* sc1, const float* sc2, const float* sc3,
                 const float* s_im_p,
                 unsigned char* __restrict__ lfrag8,
                 signed char*  __restrict__ r8t) {
    __shared__ float q0s[256];
    const int t = threadIdx.x;
    const float s0 = *sc0, s1 = *sc1, s2 = *sc2, s3 = *sc3;
    const float inv_sim = 1.0f / (*s_im_p);

    q0s[t] = fq(core0[t], s0);
    __syncthreads();

    for (int u = 0; u < 64; ++u) {
        int o = t * 64 + u;
        int i = o >> 10, j = (o >> 4) & 63, k = o & 15;
        float acc = 0.f;
        #pragma unroll
        for (int r = 0; r < 16; ++r)
            acc += q0s[i * 16 + r] * fq(core1[(r * 64 + j) * 16 + k], s1);
        int q = q8(acc, inv_sim);
        int K = i * 64 + j;
        int c = K >> 6, Kl = K & 63;
        int g = Kl >> 5, rem = Kl & 31;
        int half = rem >> 4, r2 = rem & 15;
        int v = g * 4 + (r2 >> 2), b = r2 & 3;
        int lane = half * 16 + k;
        lfrag8[(((c * 32 + lane) * 8 + v) << 2) + b] = (unsigned char)(q & 0xff);
    }

    for (int u = 0; u < 256; ++u) {
        int o = t * 256 + u;
        int i = o >> 12, col = o & 4095;
        int j = col >> 6, k = col & 63;
        float acc = 0.f;
        #pragma unroll
        for (int r = 0; r < 16; ++r)
            acc += fq(core2[(i * 64 + j) * 16 + r], s2) * fq(core3[r * 64 + k], s3);
        r8t[col * 16 + i] = (signed char)q8(acc, inv_sim);
    }
}

__global__ __launch_bounds__(256)
void gemm1_kernel(const float* __restrict__ x,
                  const unsigned int* __restrict__ lfrag,
                  signed char* __restrict__ mid8,
                  const float* s_in_p, const float* s_im_p, const float* s_x_p) {
    const int lane = threadIdx.x & 31;
    const int wave = threadIdx.x >> 5;
    const int tile = blockIdx.x * 8 + wave;
    const float inv_sin = 1.0f / (*s_in_p);
    const float cvt = (*s_in_p) * (*s_im_p);
    const float inv_sx = 1.0f / (*s_x_p);

    const int M = lane & 15, half = lane >> 4;
    const float* xrow = x + (size_t)(tile * 16 + M) * 1024;

    v8i acc = {};
    for (int c = 0; c < 16; ++c) {
        if (c + 1 < 16)
            __builtin_prefetch(xrow + (c + 1) * 64 + (half << 3), 0, 0);
        v8i a;
        #pragma unroll
        for (int v = 0; v < 8; ++v) {
            int Kb = c * 64 + ((v >> 2) << 5) + (half << 4) + ((v & 3) << 2);
            const v4f f = __builtin_nontemporal_load((const v4f*)(xrow + Kb));
            int b0 = q8(f.x, inv_sin) & 255;
            int b1 = q8(f.y, inv_sin) & 255;
            int b2 = q8(f.z, inv_sin) & 255;
            int b3 = q8(f.w, inv_sin) & 255;
            a[v] = b0 | (b1 << 8) | (b2 << 16) | (b3 << 24);
        }
        const uint4* lp = (const uint4*)(lfrag + (c * 32 + lane) * 8);
        uint4 lo = lp[0], hi = lp[1];
        v8i bfr;
        bfr[0] = (int)lo.x; bfr[1] = (int)lo.y; bfr[2] = (int)lo.z; bfr[3] = (int)lo.w;
        bfr[4] = (int)hi.x; bfr[5] = (int)hi.y; bfr[6] = (int)hi.z; bfr[7] = (int)hi.w;
        acc = __builtin_amdgcn_wmma_i32_16x16x64_iu8(true, a, true, bfr, acc,
                                                      false,  false);
    }
    const int N = lane & 15;
    const int Mbase = (lane >> 4) * 8;
    signed char* mp = mid8 + tile * 256;
    #pragma unroll
    for (int r = 0; r < 8; ++r) {
        float m = (float)acc[r] * cvt;
        mp[(Mbase + r) * 16 + N] = (signed char)q8(m, inv_sx);
    }
}

__global__ __launch_bounds__(256)
void gemm2_kernel(const signed char* __restrict__ mid8,
                  const signed char* __restrict__ r8t,
                  const float* __restrict__ bias,
                  float* __restrict__ out,
                  const float* s_x_p, const float* s_im_p) {
    const int lane = threadIdx.x & 31;
    const int wave = threadIdx.x >> 5;
    const unsigned wid = blockIdx.x * 8u + wave;
    const int mtile  = wid >> 6;
    const int ntile0 = (wid & 63) * 4;
    const float cvt = (*s_x_p) * (*s_im_p);
    const int N = lane & 15, half = lane >> 4;

    v8i a = {};
    const unsigned int* mrow =
        (const unsigned int*)(mid8 + mtile * 256 + N * 16 + (half << 3));
    a[0] = (int)mrow[0];
    a[1] = (int)mrow[1];

    v8i acc[4];
    #pragma unroll
    for (int t = 0; t < 4; ++t) {
        v8i b = {};
        {
            const uint2 rr = *(const uint2*)(r8t + ((ntile0 + t) * 16 + N) * 16 + (half << 3));
            b[0] = (int)rr.x; b[1] = (int)rr.y;
        }
        v8i z = {};
        acc[t] = __builtin_amdgcn_wmma_i32_16x16x64_iu8(true, a, true, b, z, false, false);
    }

    #pragma unroll
    for (int t = 0; t < 4; ++t) {
        const int col = (ntile0 + t) * 16 + N;
        const float bv = bias[col];
        float* orow = out + (size_t)(mtile * 16 + (half << 3)) * 4096 + col;
        #pragma unroll
        for (int r = 0; r < 8; ++r)
            __builtin_nontemporal_store((float)acc[t][r] * cvt + bv,
                                        orow + (size_t)r * 4096);
    }
}

extern "C" void kernel_launch(void* const* d_in, const int* in_sizes, int n_in,
                              void* d_out, int out_size, void* d_ws, size_t ws_size,
                              hipStream_t stream) {
    const float* x     = (const float*)d_in[0];
    const float* core0 = (const float*)d_in[1];
    const float* core1 = (const float*)d_in[2];
    const float* core2 = (const float*)d_in[3];
    const float* core3 = (const float*)d_in[4];
    const float* bias  = (const float*)d_in[5];
    const float* sc0   = (const float*)d_in[6];
    const float* sc1   = (const float*)d_in[7];
    const float* sc2   = (const float*)d_in[8];
    const float* sc3   = (const float*)d_in[9];
    const float* s_in  = (const float*)d_in[10];
    const float* s_im  = (const float*)d_in[11];
    const float* s_x   = (const float*)d_in[12];

    unsigned char* lfrag = (unsigned char*)d_ws;
    signed char*   r8t   = (signed char*)d_ws + 16384;
    signed char*   mid8  = (signed char*)d_ws + 16384 + 65536;

    prep_kernel<<<1, 256, 0, stream>>>(core0, core1, core2, core3,
                                       sc0, sc1, sc2, sc3, s_im, lfrag, r8t);
    gemm1_kernel<<<64, 256, 0, stream>>>(x, (const unsigned int*)lfrag, mid8,
                                         s_in, s_im, s_x);
    gemm2_kernel<<<4096, 256, 0, stream>>>(mid8, r8t, bias, (float*)d_out,
                                           s_x, s_im);
}
